// RWKV_TimeMix_75522704933162
// MI455X (gfx1250) — matrix operand order repaired and checked by reading, not compiled, not run on hardware
//
#include <hip/hip_runtime.h>
#include <hip/hip_bf16.h>

// RWKV TimeMix for MI455X (gfx1250).
// Four C x C GEMMs via v_wmma_f32_16x16x32_f16 with double-buffered LDS B tiles
// staged by async global->LDS copies (ASYNCcnt) when available, and 4-way
// register blocking (4 WMMAs per A fragment). f32 WKV recurrence.
// B=8, T=1024, C=1024.

#define RWKV_B 8
#define RWKV_T 1024
#define RWKV_C 1024
#define RWKV_M (RWKV_B * RWKV_T)   // 8192 rows

typedef __attribute__((ext_vector_type(16))) _Float16 v16h;
typedef __attribute__((ext_vector_type(8)))  _Float16 v8h;
typedef __attribute__((ext_vector_type(8)))  float    v8f;
typedef __attribute__((ext_vector_type(4)))  int      v4i;

union AFrag { v16h v; v8h h[2]; };

#if defined(__gfx1250__) && __has_builtin(__builtin_amdgcn_global_load_async_to_lds_b128)
#define USE_ASYNC_LDS 1
#else
#define USE_ASYNC_LDS 0
#endif

#if USE_ASYNC_LDS
// Per hipcc diagnostic: param0 is 'int4 __device__ *' (global / AS1 int4*).
typedef __attribute__((address_space(1))) v4i* as1_v4ip;
typedef __attribute__((address_space(3))) v4i* as3_v4ip;

__device__ __forceinline__ void async_b128_to_lds(const _Float16* gsrc, _Float16* lds) {
    __builtin_amdgcn_global_load_async_to_lds_b128(
        (as1_v4ip)gsrc, (as3_v4ip)lds, /*offset=*/0, /*cpol=*/0);
}

__device__ __forceinline__ void wait_asynccnt0() {
#if __has_builtin(__builtin_amdgcn_s_wait_asynccnt)
    __builtin_amdgcn_s_wait_asynccnt(0);
#else
    asm volatile("s_wait_asynccnt 0x0" ::: "memory");
#endif
}
#endif

// ---------------------------------------------------------------------------
// Cast a flat f32 array to f16.
// ---------------------------------------------------------------------------
__global__ void cast_f32_to_f16_kernel(const float* __restrict__ src,
                                       _Float16* __restrict__ dst, int n) {
    int i = blockIdx.x * blockDim.x + threadIdx.x;
    if (i < n) dst[i] = (_Float16)src[i];
}

// ---------------------------------------------------------------------------
// Time-mix: xk/xv/xr = x*mix + x_shift*(1-mix), cast to f16.
// ---------------------------------------------------------------------------
__global__ void timemix_kernel(const float* __restrict__ x,
                               const float* __restrict__ tmk,
                               const float* __restrict__ tmv,
                               const float* __restrict__ tmr,
                               _Float16* __restrict__ xk,
                               _Float16* __restrict__ xv,
                               _Float16* __restrict__ xr) {
    size_t idx = (size_t)blockIdx.x * blockDim.x + threadIdx.x;  // over B*T*C
    int c = (int)(idx % RWKV_C);
    int t = (int)((idx / RWKV_C) % RWKV_T);
    float xc = x[idx];
    float xp = (t == 0) ? 0.0f : x[idx - RWKV_C];
    float mk = tmk[c], mv = tmv[c], mr = tmr[c];
    xk[idx] = (_Float16)(xc * mk + xp * (1.0f - mk));
    xv[idx] = (_Float16)(xc * mv + xp * (1.0f - mv));
    xr[idx] = (_Float16)(xc * mr + xp * (1.0f - mr));
}

// ---------------------------------------------------------------------------
// GEMM: D[M,N] = A[M,K] * W[N,K]^T  (einsum('mk,nk->mn')).
// Block tile: 128M x 64N, 256 threads = 8 waves. Double-buffered 64x32-half
// B chunks in LDS: chunk i+1 is staged (async global->LDS when available)
// while the 4 WMMAs of chunk i execute; one barrier per K-step.
// A 16x32 and B 32x16 fragments both use contiguous k = 16*hi + s; f32 C/D per CDNA5 ISA 7.12.2.
// MODE 0: store f32.  MODE 1: sigmoid epilogue then store f32.
// ---------------------------------------------------------------------------
template <int MODE>
__global__ void __launch_bounds__(256)
gemm_xwt_wmma_kernel(const _Float16* __restrict__ A,  // [M,K]
                     const _Float16* __restrict__ W,  // [N,K]
                     float* __restrict__ D,           // [M,N]
                     int M, int N, int K) {
    __shared__ _Float16 Bs[2][64 * 32];   // 2 x 4 KB double buffer, [n][k]

    const int tid  = threadIdx.x;
    const int lane = tid & 31;
    const int wave = tid >> 5;                 // 0..7
    const int l    = lane & 15;
    const int hi   = lane >> 4;                // 0: lanes 0-15, 1: lanes 16-31

    const int n0 = blockIdx.x * 64;            // N tile base
    const int m0 = blockIdx.y * 128 + wave * 16;  // this wave's M band

    const _Float16* aRow = A + (size_t)(m0 + l) * K;

    // Cooperative B staging: thread -> (row n, 8-half column chunk), one b128
    const int bn = tid >> 2;                   // 0..63
    const int bk = (tid & 3) * 8;              // 0,8,16,24
    const _Float16* bSrc = W + (size_t)(n0 + bn) * K + bk;
    _Float16* bDst[2] = { &Bs[0][bn * 32 + bk], &Bs[1][bn * 32 + bk] };

    v8f acc0 = {}, acc1 = {}, acc2 = {}, acc3 = {};

    // ---- Prologue: stage chunk 0 into buffer 0 ----
#if USE_ASYNC_LDS
    async_b128_to_lds(bSrc, bDst[0]);
    wait_asynccnt0();
    __syncthreads();
#else
    *(v8h*)bDst[0] = *(const v8h*)bSrc;
    __syncthreads();
#endif

    for (int k0 = 0; k0 < K; k0 += 32) {
        const int buf  = (k0 >> 5) & 1;
        const bool more = (k0 + 32 < K);       // wave-uniform

        // ---- Stage next chunk into the other buffer (overlapped) ----
#if USE_ASYNC_LDS
        if (more) async_b128_to_lds(bSrc + k0 + 32, bDst[buf ^ 1]);
        v8h nextB = {};
#else
        v8h nextB = {};
        if (more) nextB = *(const v8h*)(bSrc + k0 + 32);  // load early, store late
#endif

        // A 16x32 fragment: lane holds K = {k0+16*hi..+15}
        AFrag a;
        a.v = *(const v16h*)(aRow + k0 + 16 * hi);

        if (k0 + 64 < K) {                     // prefetch A further ahead (L2)
            __builtin_prefetch(aRow + k0 + 64, 0, 1);
        }

        // Four B 32x16 fragments from LDS: lane = col N, 32 contiguous bytes
        const _Float16* BsCur = &Bs[buf][0];
        v16h b0 = *(const v16h*)&BsCur[( 0 + l) * 32 + 16 * hi];
        v16h b1 = *(const v16h*)&BsCur[(16 + l) * 32 + 16 * hi];
        v16h b2 = *(const v16h*)&BsCur[(32 + l) * 32 + 16 * hi];
        v16h b3 = *(const v16h*)&BsCur[(48 + l) * 32 + 16 * hi];

        acc0 = __builtin_amdgcn_wmma_f32_16x16x32_f16(false, a.v, false, b0,
                                                      (short)0, acc0, false, false);
        acc1 = __builtin_amdgcn_wmma_f32_16x16x32_f16(false, a.v, false, b1,
                                                      (short)0, acc1, false, false);
        acc2 = __builtin_amdgcn_wmma_f32_16x16x32_f16(false, a.v, false, b2,
                                                      (short)0, acc2, false, false);
        acc3 = __builtin_amdgcn_wmma_f32_16x16x32_f16(false, a.v, false, b3,
                                                      (short)0, acc3, false, false);

        // ---- Publish next chunk, single barrier per K-step ----
#if USE_ASYNC_LDS
        if (more) wait_asynccnt0();
        __syncthreads();
#else
        if (more) *(v8h*)bDst[buf ^ 1] = nextB;
        __syncthreads();
#endif
        (void)nextB;
    }

    // Epilogue. C/D layout: VGPR r -> row m0 + r + 8*hi, col sub-tile base + l
    float* dBase = D + (size_t)(m0 + 8 * hi) * N + n0 + l;
#pragma unroll
    for (int nt = 0; nt < 4; ++nt) {
        const v8f* acc = (nt == 0) ? &acc0 : (nt == 1) ? &acc1
                        : (nt == 2) ? &acc2 : &acc3;
        float* dCol = dBase + nt * 16;
#pragma unroll
        for (int r = 0; r < 8; ++r) {
            float val = (*acc)[r];
            if (MODE == 1) val = 1.0f / (1.0f + __expf(-val));  // sigmoid
            dCol[(size_t)r * N] = val;
        }
    }
}

// ---------------------------------------------------------------------------
// WKV recurrence (numerically stable log-sum-exp form), fused with sr * y and
// f16 cast of the result. One thread per (b, c) channel, sequential over T.
// Lanes map to consecutive c -> coalesced per-timestep accesses.
// ---------------------------------------------------------------------------
__global__ void wkv_kernel(const float* __restrict__ k,
                           const float* __restrict__ v,
                           const float* __restrict__ sr,
                           const float* __restrict__ w_param,
                           const float* __restrict__ u_param,
                           _Float16* __restrict__ a_out) {
    int idx = blockIdx.x * blockDim.x + threadIdx.x;  // over B*C
    if (idx >= RWKV_B * RWKV_C) return;
    int b = idx / RWKV_C;
    int c = idx % RWKV_C;

    const float w = -__expf(w_param[c]);
    const float u = u_param[c];

    float aa = 0.0f, bb = 0.0f, pp = -1e38f;
    size_t base = (size_t)b * RWKV_T * RWKV_C + c;
    for (int t = 0; t < RWKV_T; ++t) {
        size_t off = base + (size_t)t * RWKV_C;
        float kt = k[off];
        float vt = v[off];

        float ww = u + kt;
        float p  = fmaxf(pp, ww);
        float e1 = __expf(pp - p);
        float e2 = __expf(ww - p);
        float y  = (e1 * aa + e2 * vt) / (e1 * bb + e2);

        float ww2 = pp + w;
        float p2  = fmaxf(ww2, kt);
        float e1b = __expf(ww2 - p2);
        float e2b = __expf(kt - p2);
        aa = e1b * aa + e2b * vt;
        bb = e1b * bb + e2b;
        pp = p2;

        a_out[off] = (_Float16)(sr[off] * y);
    }
}

// ---------------------------------------------------------------------------
// Host-side launcher.
// Inputs (setup_inputs order):
//   0:x [B,T,C] f32   1:time_decay [C]   2:time_first [C]
//   3:time_mix_k [C]  4:time_mix_v [C]   5:time_mix_r [C]
//   6:Wk [C,C]        7:Wv [C,C]         8:Wr [C,C]         9:Wo [C,C]
// Output: [B,T,C] f32.
// ---------------------------------------------------------------------------
extern "C" void kernel_launch(void* const* d_in, const int* in_sizes, int n_in,
                              void* d_out, int out_size, void* d_ws, size_t ws_size,
                              hipStream_t stream) {
    const float* x   = (const float*)d_in[0];
    const float* td  = (const float*)d_in[1];
    const float* tf  = (const float*)d_in[2];
    const float* tmk = (const float*)d_in[3];
    const float* tmv = (const float*)d_in[4];
    const float* tmr = (const float*)d_in[5];
    const float* Wk  = (const float*)d_in[6];
    const float* Wv  = (const float*)d_in[7];
    const float* Wr  = (const float*)d_in[8];
    const float* Wo  = (const float*)d_in[9];
    float* out = (float*)d_out;

    const int C  = RWKV_C;
    const int M  = RWKV_M;           // B*T = 8192
    const int CC = C * C;            // 1M
    const size_t MC = (size_t)M * C; // 8M

    // Workspace layout (bytes)
    char* ws = (char*)d_ws;
    const size_t MB = 1024 * 1024;
    _Float16* Wk16 = (_Float16*)(ws + 0 * MB);    // 2 MB each
    _Float16* Wv16 = (_Float16*)(ws + 2 * MB);
    _Float16* Wr16 = (_Float16*)(ws + 4 * MB);
    _Float16* Wo16 = (_Float16*)(ws + 6 * MB);
    _Float16* xk   = (_Float16*)(ws + 8 * MB);    // 16 MB each
    _Float16* xv   = (_Float16*)(ws + 24 * MB);
    _Float16* xr   = (_Float16*)(ws + 40 * MB);
    float*    kbuf = (float*)(ws + 56 * MB);      // 32 MB each
    float*    vbuf = (float*)(ws + 88 * MB);
    float*    srb  = (float*)(ws + 120 * MB);
    _Float16* abuf = (_Float16*)(ws + 8 * MB);    // reuse xk slot (consumed by then)

    // 1) Cast weights f32 -> f16
    {
        dim3 blk(256), grd((CC + 255) / 256);
        cast_f32_to_f16_kernel<<<grd, blk, 0, stream>>>(Wk, Wk16, CC);
        cast_f32_to_f16_kernel<<<grd, blk, 0, stream>>>(Wv, Wv16, CC);
        cast_f32_to_f16_kernel<<<grd, blk, 0, stream>>>(Wr, Wr16, CC);
        cast_f32_to_f16_kernel<<<grd, blk, 0, stream>>>(Wo, Wo16, CC);
    }

    // 2) Time-mix -> xk/xv/xr (f16)
    {
        dim3 blk(256), grd((unsigned)(MC / 256));
        timemix_kernel<<<grd, blk, 0, stream>>>(x, tmk, tmv, tmr, xk, xv, xr);
    }

    // 3) Projections via WMMA GEMMs: block = 128M x 64N (8 waves)
    dim3 gblk(256);
    dim3 ggrd(C / 64, M / 128);
    gemm_xwt_wmma_kernel<0><<<ggrd, gblk, 0, stream>>>(xk, Wk16, kbuf, M, C, C);
    gemm_xwt_wmma_kernel<0><<<ggrd, gblk, 0, stream>>>(xv, Wv16, vbuf, M, C, C);
    gemm_xwt_wmma_kernel<1><<<ggrd, gblk, 0, stream>>>(xr, Wr16, srb,  M, C, C);

    // 4) WKV recurrence fused with sr*y, cast to f16
    {
        dim3 blk(256), grd((RWKV_B * RWKV_C) / 256);
        wkv_kernel<<<grd, blk, 0, stream>>>(kbuf, vbuf, srb, td, tf, abuf);
    }

    // 5) Output projection: out = abuf * Wo^T  (f32 result straight to d_out)
    gemm_xwt_wmma_kernel<0><<<ggrd, gblk, 0, stream>>>(abuf, Wo16, out, M, C, C);
}
